// QuanvolutionModule_65481071406483
// MI455X (gfx1250) — compile-verified, matrix operand order checked by reading, not run on hardware
//
#include <hip/hip_runtime.h>
#include <hip/hip_bf16.h>
#include <math.h>

// ---------- types ----------
typedef __attribute__((ext_vector_type(16))) _Float16 v16h;
typedef __attribute__((ext_vector_type(8)))  _Float16 v8h;
typedef __attribute__((ext_vector_type(2)))  _Float16 v2h;
typedef __attribute__((ext_vector_type(8)))  float    v8f;
typedef __attribute__((ext_vector_type(2)))  float    v2f;

#define B_SAMPLES   4096
#define N_PATCH     196
#define K_FEAT      784
#define K_PAD       800      // K padded to multiple of 32 for WMMA
#define NUM_CLS     10
#define N_PAD       16       // N padded to 16 (W rows 10..15 zeroed)

#define N_FLAT      (B_SAMPLES * N_PATCH)   // 802816 patches
#define N_TILES     (N_FLAT / 16)           // 50176 tiles of 16 patches
#define TILES_PER_WAVE 8
#define WAVES_B     (N_TILES / TILES_PER_WAVE)  // 6272 waves
#define BLOCKS_B    (WAVES_B / 8)               // 784 blocks of 8 waves

// workspace layout (bytes)
#define WS_OFF_M      0                      // 4*16*16 f32   = 4096 B
#define WS_OFF_WH     4096                   // 16*800 f16    = 25600 B
#define WS_OFF_FEATS  32768                  // 4096*800 f16  = 6553600 B

// =====================================================================
// Kernel A: build M_w = Re(U^dagger Z_w U), 4 x 16x16 f32, from the 12
// shared variational rotation angles.  One block, 256 threads.
// =====================================================================
__global__ __launch_bounds__(256)
void quanv_build_M(const float* __restrict__ vp, float* __restrict__ M)
{
    __shared__ float Ur[16][16];   // [k][col]
    __shared__ float Ui[16][16];
    const int t = threadIdx.x;

    if (t < 16) {
        float re[16], im[16];
        #pragma unroll
        for (int k = 0; k < 16; ++k) { re[k] = (k == t) ? 1.f : 0.f; im[k] = 0.f; }

        #pragma unroll
        for (int l = 0; l < 3; ++l) {
            #pragma unroll
            for (int w = 0; w < 4; ++w) {
                const float th = vp[l * 4 + w] * 0.5f;
                const float c = __cosf(th), s = __sinf(th);
                const int bit = 1 << (3 - w);
                #pragma unroll
                for (int i0 = 0; i0 < 16; ++i0) {
                    if (i0 & bit) continue;
                    const int i1 = i0 | bit;
                    const float ar = re[i0], ai = im[i0];
                    const float br = re[i1], bi = im[i1];
                    float nr0, ni0, nr1, ni1;
                    if (l == 0) {            // RX
                        nr0 =  c*ar + s*bi;  ni0 =  c*ai - s*br;
                        nr1 =  s*ai + c*br;  ni1 = -s*ar + c*bi;
                    } else if (l == 1) {     // RY
                        nr0 =  c*ar - s*br;  ni0 =  c*ai - s*bi;
                        nr1 =  s*ar + c*br;  ni1 =  s*ai + c*bi;
                    } else {                 // RZ
                        nr0 =  c*ar + s*ai;  ni0 =  c*ai - s*ar;
                        nr1 =  c*br - s*bi;  ni1 =  c*bi + s*br;
                    }
                    re[i0] = nr0; im[i0] = ni0;
                    re[i1] = nr1; im[i1] = ni1;
                }
            }
            #pragma unroll
            for (int w = 0; w < 3; ++w) {
                const int cb = 1 << (3 - w);
                const int tb = 1 << (3 - (w + 1));
                #pragma unroll
                for (int i = 0; i < 16; ++i) {
                    if ((i & cb) && !(i & tb)) {
                        const int j = i | tb;
                        float tr = re[i]; re[i] = re[j]; re[j] = tr;
                        float ti = im[i]; im[i] = im[j]; im[j] = ti;
                    }
                }
            }
        }
        #pragma unroll
        for (int k = 0; k < 16; ++k) { Ur[k][t] = re[k]; Ui[k][t] = im[k]; }
    }
    __syncthreads();

    const int i = t >> 4, j = t & 15;
    float acc0 = 0.f, acc1 = 0.f, acc2 = 0.f, acc3 = 0.f;
    #pragma unroll
    for (int k = 0; k < 16; ++k) {
        const float prod = Ur[k][i] * Ur[k][j] + Ui[k][i] * Ui[k][j];
        acc0 += (k & 8) ? -prod : prod;
        acc1 += (k & 4) ? -prod : prod;
        acc2 += (k & 2) ? -prod : prod;
        acc3 += (k & 1) ? -prod : prod;
    }
    M[0 * 256 + i * 16 + j] = acc0;
    M[1 * 256 + i * 16 + j] = acc1;
    M[2 * 256 + i * 16 + j] = acc2;
    M[3 * 256 + i * 16 + j] = acc3;
}

// =====================================================================
// Kernel A2: padded f16 weights Wh[16][800] (rows>=10, cols>=784 zero)
// and zero the 16 pad feature columns of every sample row.
// =====================================================================
__global__ __launch_bounds__(256)
void quanv_prep_W(const float* __restrict__ W, _Float16* __restrict__ Wh)
{
    const int g = blockIdx.x * 256 + threadIdx.x;
    if (g >= N_PAD * K_PAD) return;
    const int n = g / K_PAD, k = g % K_PAD;
    const float val = (n < NUM_CLS && k < K_FEAT) ? W[n * K_FEAT + k] : 0.f;
    Wh[g] = (_Float16)val;
}

__global__ __launch_bounds__(256)
void quanv_zero_pad(_Float16* __restrict__ feats)
{
    const int g = blockIdx.x * 256 + threadIdx.x;   // < 4096*16
    const int b = g >> 4, k = K_FEAT + (g & 15);
    feats[(size_t)b * K_PAD + k] = (_Float16)0.f;
}

// =====================================================================
// Kernel B: quadratic forms on the matrix pipe.  Each wave32 owns 8 tiles (one per iteration)
// of 16 flat patches; V (16 patch-vectors x 16 states, f32) multiplies
// the four M_w via V_WMMA_F32_16X16X4_F32 (4 K-chunks of 4 per wire).
// f_w[p] = <P_w[p,:], V[p,:]> finished with an LDS transpose + 16 FMA.
// The Kron vector is kept in 16 NAMED scalars (no private array!) so
// lane-dependent picks lower to single v_cndmask, not 16-way chains.
// =====================================================================
__global__ __launch_bounds__(256)
void quanv_feats_wmma(const float* __restrict__ x,
                      const float* __restrict__ M4,
                      _Float16* __restrict__ feats)
{
    // per-wave scratch: 4 tiles of 16x16 f32, row stride 17 (conflict-free)
    __shared__ float Pbuf[8][4][16][17];

    const int lane = threadIdx.x & 31;
    const int wv   = threadIdx.x >> 5;
    const int r    = lane & 15;                 // row / col selector
    const bool lo  = (lane < 16);
    const int kofs = lo ? 0 : 2;                // K offset within a chunk of 4
    const int rbase = lo ? 0 : 8;               // C/D row base

    // ---- preload B fragments of the 4 M_w (uniform across tiles) ----
    v2f Bf[4][4];
    #pragma unroll
    for (int w = 0; w < 4; ++w)
        #pragma unroll
        for (int c = 0; c < 4; ++c) {
            const int k = 4 * c + kofs;         // address math only
            Bf[w][c].x = M4[w * 256 + k * 16 + r];
            Bf[w][c].y = M4[w * 256 + (k + 1) * 16 + r];
        }

    const int tile0 = (blockIdx.x * 8 + wv) * TILES_PER_WAVE;

    #pragma unroll 1
    for (int it = 0; it < TILES_PER_WAVE; ++it) {
        const int tile = tile0 + it;
        const int pidx = tile * 16 + r;         // flat (sample,patch) id
        const int b = pidx / N_PATCH;
        const int p = pidx - b * N_PATCH;
        const int pr = p / 14, pc = p - pr * 14;

        // ---- encoder vector v = kron of (cos t/2, sin t/2), named scalars ----
        const float* px = x + (size_t)b * 784 + (2 * pr) * 28 + 2 * pc;
        const float t0 = px[0] * 0.5f, t1 = px[1] * 0.5f;
        const float t2 = px[28] * 0.5f, t3 = px[29] * 0.5f;
        const float c0 = __cosf(t0), s0 = __sinf(t0);
        const float c1 = __cosf(t1), s1 = __sinf(t1);
        const float c2 = __cosf(t2), s2 = __sinf(t2);
        const float c3 = __cosf(t3), s3 = __sinf(t3);

        const float cc01 = c0 * c1, cs01 = c0 * s1, sc01 = s0 * c1, ss01 = s0 * s1;
        const float cc23 = c2 * c3, cs23 = c2 * s3, sc23 = s2 * c3, ss23 = s2 * s3;

        // i = w0*8 + w1*4 + w2*2 + w3 (bit 0 -> cos, bit 1 -> sin)
        const float v0  = cc01 * cc23, v1  = cc01 * cs23, v2  = cc01 * sc23, v3  = cc01 * ss23;
        const float v4  = cs01 * cc23, v5  = cs01 * cs23, v6  = cs01 * sc23, v7  = cs01 * ss23;
        const float v8  = sc01 * cc23, v9  = sc01 * cs23, v10 = sc01 * sc23, v11 = sc01 * ss23;
        const float v12 = ss01 * cc23, v13 = ss01 * cs23, v14 = ss01 * sc23, v15 = ss01 * ss23;

        // ---- A fragments: single value-select per element ----
        v2f af0, af1, af2, af3;
        af0.x = lo ? v0  : v2 ;  af0.y = lo ? v1  : v3 ;
        af1.x = lo ? v4  : v6 ;  af1.y = lo ? v5  : v7 ;
        af2.x = lo ? v8  : v10;  af2.y = lo ? v9  : v11;
        af3.x = lo ? v12 : v14;  af3.y = lo ? v13 : v15;

        // ---- P_w = V * M_w via f32 WMMA, K accumulated in chunks of 4 ----
        v8f P0 = {}, P1 = {}, P2 = {}, P3 = {};
        P0 = __builtin_amdgcn_wmma_f32_16x16x4_f32(false, af0, false, Bf[0][0], (short)0, P0, false, false);
        P1 = __builtin_amdgcn_wmma_f32_16x16x4_f32(false, af0, false, Bf[1][0], (short)0, P1, false, false);
        P2 = __builtin_amdgcn_wmma_f32_16x16x4_f32(false, af0, false, Bf[2][0], (short)0, P2, false, false);
        P3 = __builtin_amdgcn_wmma_f32_16x16x4_f32(false, af0, false, Bf[3][0], (short)0, P3, false, false);
        P0 = __builtin_amdgcn_wmma_f32_16x16x4_f32(false, af1, false, Bf[0][1], (short)0, P0, false, false);
        P1 = __builtin_amdgcn_wmma_f32_16x16x4_f32(false, af1, false, Bf[1][1], (short)0, P1, false, false);
        P2 = __builtin_amdgcn_wmma_f32_16x16x4_f32(false, af1, false, Bf[2][1], (short)0, P2, false, false);
        P3 = __builtin_amdgcn_wmma_f32_16x16x4_f32(false, af1, false, Bf[3][1], (short)0, P3, false, false);
        P0 = __builtin_amdgcn_wmma_f32_16x16x4_f32(false, af2, false, Bf[0][2], (short)0, P0, false, false);
        P1 = __builtin_amdgcn_wmma_f32_16x16x4_f32(false, af2, false, Bf[1][2], (short)0, P1, false, false);
        P2 = __builtin_amdgcn_wmma_f32_16x16x4_f32(false, af2, false, Bf[2][2], (short)0, P2, false, false);
        P3 = __builtin_amdgcn_wmma_f32_16x16x4_f32(false, af2, false, Bf[3][2], (short)0, P3, false, false);
        P0 = __builtin_amdgcn_wmma_f32_16x16x4_f32(false, af3, false, Bf[0][3], (short)0, P0, false, false);
        P1 = __builtin_amdgcn_wmma_f32_16x16x4_f32(false, af3, false, Bf[1][3], (short)0, P1, false, false);
        P2 = __builtin_amdgcn_wmma_f32_16x16x4_f32(false, af3, false, Bf[2][3], (short)0, P2, false, false);
        P3 = __builtin_amdgcn_wmma_f32_16x16x4_f32(false, af3, false, Bf[3][3], (short)0, P3, false, false);

        // ---- scatter P tiles: lane holds col r, rows e+rbase ----
        #pragma unroll
        for (int e = 0; e < 8; ++e) {
            Pbuf[wv][0][e + rbase][r] = P0[e];
            Pbuf[wv][1][e + rbase][r] = P1[e];
            Pbuf[wv][2][e + rbase][r] = P2[e];
            Pbuf[wv][3][e + rbase][r] = P3[e];
        }

        // ---- dot with own v: lanes 0-15 do w=0,1; lanes 16-31 do w=2,3 ----
        const int w0 = lo ? 0 : 2;              // LDS address math only
        const float* Pr0 = &Pbuf[wv][w0][r][0];
        const float* Pr1 = &Pbuf[wv][w0 + 1][r][0];
        #define QDOT16(P) (P[0]*v0 + P[1]*v1 + P[2]*v2 + P[3]*v3 + \
                           P[4]*v4 + P[5]*v5 + P[6]*v6 + P[7]*v7 + \
                           P[8]*v8 + P[9]*v9 + P[10]*v10 + P[11]*v11 + \
                           P[12]*v12 + P[13]*v13 + P[14]*v14 + P[15]*v15)
        const float f0 = QDOT16(Pr0);
        const float f1 = QDOT16(Pr1);
        #undef QDOT16

        // lane r -> (w0, w0+1) of patch pidx; packed 2xf16 store (4B aligned)
        v2h hv; hv.x = (_Float16)f0; hv.y = (_Float16)f1;
        *(v2h*)(feats + (size_t)b * K_PAD + p * 4 + w0) = hv;
    }
}

// =====================================================================
// Kernel C: classifier GEMM via v_wmma_f32_16x16x32_f16 + fused bias
// and log-softmax.  One wave32 per 16-row tile; K loop = 800/32 = 25.
// =====================================================================
__global__ __launch_bounds__(32)
void quanv_classify(const _Float16* __restrict__ feats,
                    const _Float16* __restrict__ Wh,
                    const float* __restrict__ bias,
                    float* __restrict__ out)
{
    __shared__ float tile[16][16];

    const int lane = threadIdx.x;
    const int m0   = blockIdx.x * 16;
    const int rsel = lane & 15;
    const int kb   = (lane < 16) ? 0 : 8;

    const _Float16* arow = feats + (size_t)(m0 + rsel) * K_PAD;
    const _Float16* brow = Wh + (size_t)rsel * K_PAD;

    v8f acc = {};
    #pragma unroll 1
    for (int kk = 0; kk < 25; ++kk) {
        const int k0 = kk * 32;
        const v8h a0 = *(const v8h*)(arow + k0 + kb);
        const v8h a1 = *(const v8h*)(arow + k0 + 16 + kb);
        const v8h b0 = *(const v8h*)(brow + k0 + kb);
        const v8h b1 = *(const v8h*)(brow + k0 + 16 + kb);
        v16h Af, Bf;
        #pragma unroll
        for (int i = 0; i < 8; ++i) {
            Af[i] = a0[i]; Af[i + 8] = a1[i];
            Bf[i] = b0[i]; Bf[i + 8] = b1[i];
        }
        acc = __builtin_amdgcn_wmma_f32_16x16x32_f16(false, Af, false, Bf,
                                                     (short)0, acc, false, false);
    }

    const int rbase = (lane < 16) ? 0 : 8;
    #pragma unroll
    for (int v = 0; v < 8; ++v) tile[rbase + v][rsel] = acc[v];
    __syncthreads();

    if (lane < 16) {
        const int m = m0 + lane;
        float vals[NUM_CLS];
        float mx = -INFINITY;
        #pragma unroll
        for (int n = 0; n < NUM_CLS; ++n) {
            vals[n] = tile[lane][n] + bias[n];
            mx = fmaxf(mx, vals[n]);
        }
        float sum = 0.f;
        #pragma unroll
        for (int n = 0; n < NUM_CLS; ++n) sum += __expf(vals[n] - mx);
        const float lz = mx + __logf(sum);
        #pragma unroll
        for (int n = 0; n < NUM_CLS; ++n) out[m * NUM_CLS + n] = vals[n] - lz;
    }
}

// =====================================================================
extern "C" void kernel_launch(void* const* d_in, const int* in_sizes, int n_in,
                              void* d_out, int out_size, void* d_ws, size_t ws_size,
                              hipStream_t stream)
{
    const float* x  = (const float*)d_in[0];   // 4096*784
    const float* vp = (const float*)d_in[1];   // 3*4
    const float* W  = (const float*)d_in[2];   // 10*784
    const float* b  = (const float*)d_in[3];   // 10
    float* out      = (float*)d_out;           // 4096*10

    char* ws = (char*)d_ws;
    float*    M     = (float*)(ws + WS_OFF_M);
    _Float16* Wh    = (_Float16*)(ws + WS_OFF_WH);
    _Float16* feats = (_Float16*)(ws + WS_OFF_FEATS);

    quanv_build_M<<<1, 256, 0, stream>>>(vp, M);
    quanv_prep_W<<<(N_PAD * K_PAD + 255) / 256, 256, 0, stream>>>(W, Wh);
    quanv_zero_pad<<<(B_SAMPLES * 16) / 256, 256, 0, stream>>>(feats);
    quanv_feats_wmma<<<BLOCKS_B, 256, 0, stream>>>(x, M, feats);
    quanv_classify<<<B_SAMPLES / 16, 32, 0, stream>>>(feats, Wh, b, out);
}
